// GaussianFusionNet_35098472742934
// MI455X (gfx1250) — matrix operand order repaired and checked by reading, not compiled, not run on hardware
//
#include <hip/hip_runtime.h>
#include <hip/hip_bf16.h>
#include <math.h>

#define B_   4
#define N_   2048
#define V_   4
#define C_   64
#define H_   32
#define W_   32
#define D_   128
#define K_   64
#define OUT_ 256
#define BN_  (B_ * N_)

typedef __attribute__((ext_vector_type(16))) _Float16 v16h;
typedef __attribute__((ext_vector_type(8)))  _Float16 v8h;
typedef __attribute__((ext_vector_type(8)))  float    v8f;
typedef __attribute__((ext_vector_type(4)))  float    v4f;

__device__ __forceinline__ v8f wmma_f16(v16h a, v16h b, v8f c) {
  return __builtin_amdgcn_wmma_f32_16x16x32_f16(false, a, false, b, (short)0, c, false, false);
}

__device__ __forceinline__ v16h cat8(v8h lo, v8h hi) {
  return __builtin_shufflevector(lo, hi, 0, 1, 2, 3, 4, 5, 6, 7, 8, 9, 10, 11, 12, 13, 14, 15);
}

__device__ __forceinline__ float blk_sum(float v, float* red) {
  int t = threadIdx.x;
  red[t] = v; __syncthreads();
  for (int s = blockDim.x >> 1; s > 0; s >>= 1) {
    if (t < s) red[t] += red[t + s];
    __syncthreads();
  }
  float r = red[0]; __syncthreads();
  return r;
}

__device__ __forceinline__ float blk_max(float v, float* red) {
  int t = threadIdx.x;
  red[t] = v; __syncthreads();
  for (int s = blockDim.x >> 1; s > 0; s >>= 1) {
    if (t < s) red[t] = fmaxf(red[t], red[t + s]);
    __syncthreads();
  }
  float r = red[0]; __syncthreads();
  return r;
}

// ---------------------------------------------------------------------------
// Pack [rows,14] -> [rows,32] zero padded (for K=32 guard-free WMMA)
// ---------------------------------------------------------------------------
__global__ void k_pack14(const float* __restrict__ src, float* __restrict__ dst) {
  int r = blockIdx.x, t = threadIdx.x;
  dst[(size_t)r * 32 + t] = (t < 14) ? src[(size_t)r * 14 + t] : 0.f;
}

__global__ void k_zero(float* __restrict__ p, int n) {
  int i = blockIdx.x * 256 + threadIdx.x;
  if (i < n) p[i] = 0.f;
}

// ---------------------------------------------------------------------------
// Projection + bilinear sampling -> vfeat [BN, C]
// ---------------------------------------------------------------------------
__global__ void k_proj(const float* __restrict__ ga, const float* __restrict__ vf,
                       const float* __restrict__ ex, const float* __restrict__ intr,
                       const float* __restrict__ off, float* __restrict__ vfeat) {
  int bn = blockIdx.x;
  int b = bn >> 11;
  int c = threadIdx.x;
  const float px = ga[bn * 14 + 0], py = ga[bn * 14 + 1], pz = ga[bn * 14 + 2];
  const float ox = off[bn * 2 + 0], oy = off[bn * 2 + 1];
  float num = 0.f, den = 0.f;
  for (int v = 0; v < V_; ++v) {
    const float* E = ex + (size_t)(b * V_ + v) * 16;
    float c0 = E[0] * px + E[1] * py + E[2]  * pz + E[3];
    float c1 = E[4] * px + E[5] * py + E[6]  * pz + E[7];
    float c2 = E[8] * px + E[9] * py + E[10] * pz + E[11];
    const float* Km = intr + (size_t)(b * V_ + v) * 9;
    float u  = Km[0] * c0 + Km[1] * c1 + Km[2] * c2;
    float vv = Km[3] * c0 + Km[4] * c1 + Km[5] * c2;
    float z  = Km[6] * c0 + Km[7] * c1 + Km[8] * c2;
    float dep = fmaxf(z, 1e-6f);
    float gx = 2.f * (u  / dep) / 448.f - 1.f + ox;
    float gy = 2.f * (vv / dep) / 448.f - 1.f + oy;
    float valid = (dep > 0.1f && gx >= -1.f && gx <= 1.f && gy >= -1.f && gy <= 1.f) ? 1.f : 0.f;
    float x = (gx + 1.f) * 0.5f * (float)W_ - 0.5f;
    float y = (gy + 1.f) * 0.5f * (float)H_ - 0.5f;
    float x0 = floorf(x), y0 = floorf(y);
    const float* fmc = vf + ((size_t)(b * V_ + v) * C_ + c) * (H_ * W_);
    float s = 0.f;
    for (int dy = 0; dy < 2; ++dy)
      for (int dx = 0; dx < 2; ++dx) {
        float xi = x0 + (float)dx, yi = y0 + (float)dy;
        float wgt = (1.f - fabsf(x - xi)) * (1.f - fabsf(y - yi));
        float inb = (xi >= 0.f && xi < (float)W_ && yi >= 0.f && yi < (float)H_) ? 1.f : 0.f;
        int xc = (int)fminf(fmaxf(xi, 0.f), (float)(W_ - 1));
        int yc = (int)fminf(fmaxf(yi, 0.f), (float)(H_ - 1));
        s += fmc[yc * W_ + xc] * wgt * inb;
      }
    num += s * valid; den += valid;
  }
  vfeat[(size_t)bn * C_ + c] = num / (den + 1e-8f);
}

// ---------------------------------------------------------------------------
// Guard-free WMMA GEMM: Out = act(A[M x Kd] @ W^T + bias), W = [Nd x Kd] f32.
// M mult of 16 (grid.y = M/16), Kd mult of 32, lda/ldw mult of 4.
// ---------------------------------------------------------------------------
template <typename TA, typename TO>
__global__ __launch_bounds__(32) void k_gemm_t(const TA* __restrict__ A,
                                               const float* __restrict__ Wm,
                                               const float* __restrict__ bias,
                                               TO* __restrict__ Out,
                                               int Kd, int lda, int ldw, int ldo, int act) {
  int lane = threadIdx.x, mr = lane & 15, g = lane >> 4;
  int n0 = blockIdx.x * 16, m0 = blockIdx.y * 16;
  int arow = m0 + mr, wrow = n0 + mr;
  v8f c = {};
  for (int k0 = 0; k0 < Kd; k0 += 32) {
    v16h a, bf;
    if constexpr (__is_same(TA, float)) {
      const float* ap = A + (size_t)arow * lda + k0;
      v4f a0 = *(const v4f*)(ap + g * 8);
      v4f a1 = *(const v4f*)(ap + g * 8 + 4);
      v4f a2 = *(const v4f*)(ap + 16 + g * 8);
      v4f a3 = *(const v4f*)(ap + 16 + g * 8 + 4);
#pragma unroll
      for (int j = 0; j < 4; ++j) {
        a[j]      = (_Float16)a0[j];
        a[4 + j]  = (_Float16)a1[j];
        a[8 + j]  = (_Float16)a2[j];
        a[12 + j] = (_Float16)a3[j];
      }
    } else {
      const _Float16* ap = (const _Float16*)A + (size_t)arow * lda + k0;
      v8h lo = *(const v8h*)(ap + g * 8);
      v8h hi = *(const v8h*)(ap + 16 + g * 8);
      a = cat8(lo, hi);
    }
    {
      const float* wp = Wm + (size_t)wrow * ldw + k0;
      v4f w0 = *(const v4f*)(wp + g * 8);
      v4f w1 = *(const v4f*)(wp + g * 8 + 4);
      v4f w2 = *(const v4f*)(wp + 16 + g * 8);
      v4f w3 = *(const v4f*)(wp + 16 + g * 8 + 4);
#pragma unroll
      for (int j = 0; j < 4; ++j) {
        bf[j]      = (_Float16)w0[j];
        bf[4 + j]  = (_Float16)w1[j];
        bf[8 + j]  = (_Float16)w2[j];
        bf[12 + j] = (_Float16)w3[j];
      }
    }
    c = wmma_f16(a, bf, c);
  }
  float bv = bias ? bias[wrow] : 0.f;
#pragma unroll
  for (int r = 0; r < 8; ++r) {
    int m = m0 + r + 8 * g;
    float y = c[r] + bv;
    if (act == 1) y = 1.f / (1.f + __expf(-y));
    Out[(size_t)m * ldo + (n0 + mr)] = (TO)y;
  }
}

// ---------------------------------------------------------------------------
// Flash attention (transposed-score formulation), f16 operands, dh = 32.
// One wave per 16-query tile per (b,h).
//   S^T = K_tile(16x32) @ Q^T(32x16)  -> per-query stats live in 2 lanes
//   O^T = V^T(16x32) @ P^T(32x16)     -> vectorized output stores
// Qh: [BN,128] halves, KVh: [BN,256] halves (K: 0..127, V: 128..255)
// ---------------------------------------------------------------------------
__global__ __launch_bounds__(32) void k_flash(const _Float16* __restrict__ Qh,
                                              const _Float16* __restrict__ KVh,
                                              _Float16* __restrict__ Oh) {
  __shared__ __align__(16) _Float16 pt[16 * 32];   // P^T stored as [query][key]
  int lane = threadIdx.x, mr = lane & 15, g = lane >> 4;
  int qt = blockIdx.x;
  int bh = blockIdx.y, b = bh >> 2, h = bh & 3;
  const float scale = 0.17677669529663687f;  // 1/sqrt(32)

  // B operand: Q^T (32dh x 16q). lane col = query mr, halves j -> dh = g*8+j (j<8), 16+g*8+(j-8) (j>=8)
  const _Float16* Qp = Qh + (size_t)(b * N_ + qt * 16 + mr) * D_ + h * 32;
  v16h qb = cat8(*(const v8h*)(Qp + g * 8), *(const v8h*)(Qp + 16 + g * 8));

  float mq = -1e30f, lq = 0.f;   // per-query running max / sum (lane scalar)
  v8f o0 = {}, o1 = {};          // O^T tiles: rows dh 0..15 / 16..31, col = query mr

  for (int kt = 0; kt < N_; kt += 32) {
    const _Float16* Kp = KVh + (size_t)(b * N_ + kt) * (2 * D_) + h * 32;
    const _Float16* Vp = Kp + D_;
    // A operands: key tiles (16keys x 32dh), A-layout row = key mr
    const _Float16* kr0 = Kp + (size_t)mr * (2 * D_);
    const _Float16* kr1 = Kp + (size_t)(16 + mr) * (2 * D_);
    v16h a0 = cat8(*(const v8h*)(kr0 + g * 8), *(const v8h*)(kr0 + 16 + g * 8));
    v16h a1 = cat8(*(const v8h*)(kr1 + g * 8), *(const v8h*)(kr1 + 16 + g * 8));
    v8f c0 = {}, c1 = {};
    c0 = wmma_f16(a0, qb, c0);   // S^T keys 0..15  x queries
    c1 = wmma_f16(a1, qb, c1);   // S^T keys 16..31 x queries
    // per-query online softmax: in-lane reduce + one xor-16 exchange
    float tmax = -1e30f;
#pragma unroll
    for (int r = 0; r < 8; ++r) {
      c0[r] *= scale; c1[r] *= scale;
      tmax = fmaxf(tmax, fmaxf(c0[r], c1[r]));
    }
    tmax = fmaxf(tmax, __shfl_xor(tmax, 16, 32));
    float mn = fmaxf(mq, tmax);
    float fr = __expf(mq - mn);
    mq = mn;
    float rs = 0.f;
#pragma unroll
    for (int r = 0; r < 8; ++r) {
      float p0 = __expf(c0[r] - mn);
      float p1 = __expf(c1[r] - mn);
      rs += p0 + p1;
      pt[mr * 32 + r + 8 * g]      = (_Float16)p0;   // key r+8g
      pt[mr * 32 + 16 + r + 8 * g] = (_Float16)p1;   // key 16+r+8g
    }
    rs += __shfl_xor(rs, 16, 32);
    lq = lq * fr + rs;
#pragma unroll
    for (int r = 0; r < 8; ++r) { o0[r] *= fr; o1[r] *= fr; }
    __syncthreads();
    // B operand: P^T (32keys x 16q): contiguous keys g*16..g*16+15 for query mr
    v16h pb = cat8(*(const v8h*)(pt + mr * 32 + g * 16),
                   *(const v8h*)(pt + mr * 32 + g * 16 + 8));
    // A operands: V^T tiles (16dh x 32keys): A(m=dh, kk=key) = V[key][dh]
    v16h av0, av1;
#pragma unroll
    for (int j = 0; j < 16; ++j) {
      int kk = g * 16 + j;
      av0[j] = Vp[(size_t)kk * (2 * D_) + mr];
      av1[j] = Vp[(size_t)kk * (2 * D_) + 16 + mr];
    }
    o0 = wmma_f16(av0, pb, o0);
    o1 = wmma_f16(av1, pb, o1);
    __syncthreads();
  }
  float il = 1.f / fmaxf(lq, 1e-20f);
  _Float16* Op = Oh + (size_t)(b * N_ + qt * 16 + mr) * D_ + h * 32;
  v8h h0, h1;
#pragma unroll
  for (int r = 0; r < 8; ++r) {
    h0[r] = (_Float16)(o0[r] * il);
    h1[r] = (_Float16)(o1[r] * il);
  }
  *(v8h*)(Op + 8 * g)      = h0;   // dh 8g .. 8g+7
  *(v8h*)(Op + 16 + 8 * g) = h1;   // dh 16+8g .. 16+8g+7
}

// ---------------------------------------------------------------------------
// NetVLAD aggregation: vlad[b,k,d] = sum_n assign[b,n,k] * xn[b,n,d]
// ---------------------------------------------------------------------------
__global__ __launch_bounds__(32) void k_vlad_gemm(const float* __restrict__ As,
                                                  const float* __restrict__ Xn,
                                                  float* __restrict__ Vl) {
  int lane = threadIdx.x, mr = lane & 15, g = lane >> 4;
  int d0 = blockIdx.x * 16, k0c = blockIdx.y * 16, b = blockIdx.z;
  const float* Ab = As + (size_t)b * N_ * K_;
  const float* Xb = Xn + (size_t)b * N_ * D_;
  v8f c = {};
  for (int t0 = 0; t0 < N_; t0 += 32) {
    v16h a, bf;
#pragma unroll
    for (int j = 0; j < 16; ++j) {
      int kk = g * 16 + j;
      a[j] = (_Float16)Ab[(size_t)(t0 + kk) * K_ + (k0c + mr)];
    }
#pragma unroll
    for (int j = 0; j < 16; ++j) {
      int kk = g * 16 + j;
      bf[j] = (_Float16)Xb[(size_t)(t0 + kk) * D_ + (d0 + mr)];
    }
    c = wmma_f16(a, bf, c);
  }
#pragma unroll
  for (int r = 0; r < 8; ++r) {
    int kcl = k0c + r + 8 * g;
    Vl[((size_t)b * K_ + kcl) * D_ + d0 + mr] = c[r];
  }
}

// ---------------------------------------------------------------------------
// Elementwise / reduction kernels
// ---------------------------------------------------------------------------
__global__ void k_ln_act(const float* __restrict__ in, const float* __restrict__ res,
                         const float* __restrict__ ga, const float* __restrict__ be,
                         float* __restrict__ out, int ldin, int ldres, int ldout, int relu) {
  __shared__ float red[256];
  int row = blockIdx.x, t = threadIdx.x;
  float wdt = (float)blockDim.x;
  float x = in[(size_t)row * ldin + t];
  if (res) x += res[(size_t)row * ldres + t];
  float m = blk_sum(x, red) / wdt;
  float d = x - m;
  float v = blk_sum(d * d, red) / wdt;
  float y = d * rsqrtf(v + 1e-5f) * ga[t] + be[t];
  if (relu) y = fmaxf(y, 0.f);
  out[(size_t)row * ldout + t] = y;
}

__global__ void k_l2rows(const float* __restrict__ in, float* __restrict__ out) {
  __shared__ float red[256];
  int row = blockIdx.x, t = threadIdx.x;
  float x = in[(size_t)row * D_ + t];
  float ss = blk_sum(x * x, red);
  out[(size_t)row * D_ + t] = x / fmaxf(sqrtf(ss), 1e-12f);
}

__global__ void k_softmax64(float* __restrict__ a) {
  __shared__ float red[256];
  int row = blockIdx.x, t = threadIdx.x;
  float x = a[(size_t)row * K_ + t];
  float mx = blk_max(x, red);
  float e = __expf(x - mx);
  float s = blk_sum(e, red);
  a[(size_t)row * K_ + t] = e / s;
}

__global__ void k_gate_fuse(const float* __restrict__ fcat, const float* __restrict__ gate,
                            float* __restrict__ fused) {
  int row = blockIdx.x, t = threadIdx.x;
  float gv = gate[(size_t)row * D_ + t];
  float a = fcat[(size_t)row * (2 * D_) + t];
  float b = fcat[(size_t)row * (2 * D_) + D_ + t];
  fused[(size_t)row * D_ + t] = a * gv + b * (1.f - gv);
}

// two-stage deterministic mean over N (chunked)
__global__ void k_rowmean_p1(const float* __restrict__ g, const float* __restrict__ v,
                             float* __restrict__ part) {
  int b = blockIdx.x, ch = blockIdx.y, wh = blockIdx.z, t = threadIdx.x;
  const float* s = wh ? v : g;
  float acc = 0.f;
  int n0 = ch * (N_ / 16);
  for (int n = n0; n < n0 + (N_ / 16); ++n) acc += s[((size_t)b * N_ + n) * D_ + t];
  part[(((size_t)b * 2 + wh) * 16 + ch) * D_ + t] = acc;
}

__global__ void k_rowmean_p2(const float* __restrict__ part, float* __restrict__ ri) {
  int b = blockIdx.x, wh = blockIdx.y, t = threadIdx.x;
  float acc = 0.f;
  for (int ch = 0; ch < 16; ++ch)
    acc += part[(((size_t)b * 2 + wh) * 16 + ch) * D_ + t];
  ri[(size_t)b * (2 * D_) + wh * D_ + t] = acc * (1.f / (float)N_);
}

__global__ void k_router(const float* __restrict__ rh, const float* __restrict__ w2,
                         const float* __restrict__ b2, float* __restrict__ rwv,
                         float* __restrict__ dout) {
  int t = threadIdx.x;
  if (t < B_) {
    float l[3];
    for (int i = 0; i < 3; ++i) {
      float acc = b2[i];
      for (int d = 0; d < D_; ++d) acc += rh[t * D_ + d] * w2[i * D_ + d];
      l[i] = acc;
    }
    float mx = fmaxf(l[0], fmaxf(l[1], l[2]));
    float e0 = __expf(l[0] - mx), e1 = __expf(l[1] - mx), e2 = __expf(l[2] - mx);
    float s = e0 + e1 + e2;
    rwv[t * 3 + 0] = e0 / s; rwv[t * 3 + 1] = e1 / s; rwv[t * 3 + 2] = e2 / s;
    dout[B_ * OUT_ + t * 3 + 0] = e0 / s;
    dout[B_ * OUT_ + t * 3 + 1] = e1 / s;
    dout[B_ * OUT_ + t * 3 + 2] = e2 / s;
  }
}

// two-stage deterministic sum over N of assign -> asum [B,K]
__global__ void k_asum_p1(const float* __restrict__ assg, float* __restrict__ part) {
  int b = blockIdx.x, ch = blockIdx.y, t = threadIdx.x;
  float s = 0.f;
  int n0 = ch * (N_ / 32);
  for (int n = n0; n < n0 + (N_ / 32); ++n) s += assg[((size_t)b * N_ + n) * K_ + t];
  part[((size_t)b * 32 + ch) * K_ + t] = s;
}

__global__ void k_asum_p2(const float* __restrict__ part, float* __restrict__ asum) {
  int b = blockIdx.x, t = threadIdx.x;
  float s = 0.f;
  for (int ch = 0; ch < 32; ++ch) s += part[((size_t)b * 32 + ch) * K_ + t];
  asum[b * K_ + t] = s;
}

__global__ void k_vlad_post1(const float* __restrict__ vl, const float* __restrict__ asum,
                             const float* __restrict__ cent, float* __restrict__ desc) {
  __shared__ float red[256];
  int bk = blockIdx.x, t = threadIdx.x;
  int b = bk >> 6, k = bk & 63;
  float v = vl[(size_t)bk * D_ + t] - asum[b * K_ + k] * cent[(size_t)k * D_ + t];
  float ss = blk_sum(v * v, red);
  desc[(size_t)b * (K_ * D_) + k * D_ + t] = v / fmaxf(sqrtf(ss), 1e-12f);
}

__global__ void k_vlad_post2(float* __restrict__ desc) {
  __shared__ float red[256];
  int b = blockIdx.x, t = threadIdx.x;
  float ss = 0.f;
  for (int j = 0; j < (K_ * D_) / 256; ++j) {
    float x = desc[(size_t)b * (K_ * D_) + j * 256 + t];
    ss += x * x;
  }
  float tot = blk_sum(ss, red);
  float inv = 1.f / fmaxf(sqrtf(tot), 1e-12f);
  for (int j = 0; j < (K_ * D_) / 256; ++j)
    desc[(size_t)b * (K_ * D_) + j * 256 + t] *= inv;
}

__global__ void k_bn_post(const float* __restrict__ d2, const float* __restrict__ lg,
                          const float* __restrict__ lb, float* __restrict__ out) {
  __shared__ float red[256];
  int b = blockIdx.x, t = threadIdx.x;
  float x = d2[b * OUT_ + t];
  float m = blk_sum(x, red) / (float)OUT_;
  float dv = x - m;
  float var = blk_sum(dv * dv, red) / (float)OUT_;
  float y = dv * rsqrtf(var + 1e-5f) * lg[t] + lb[t];
  float ss = blk_sum(y * y, red);
  out[b * OUT_ + t] = y / fmaxf(sqrtf(ss), 1e-12f);
}

__global__ void k_final(const float* __restrict__ descs, const float* __restrict__ rwv,
                        float* __restrict__ dout) {
  __shared__ float red[256];
  int b = blockIdx.x, t = threadIdx.x;
  float acc = rwv[b * 3 + 0] * descs[0 * B_ * OUT_ + b * OUT_ + t]
            + rwv[b * 3 + 1] * descs[1 * B_ * OUT_ + b * OUT_ + t]
            + rwv[b * 3 + 2] * descs[2 * B_ * OUT_ + b * OUT_ + t];
  float ss = blk_sum(acc * acc, red);
  dout[b * OUT_ + t] = acc / fmaxf(sqrtf(ss), 1e-12f);
}

// ---------------------------------------------------------------------------
extern "C" void kernel_launch(void* const* d_in, const int* in_sizes, int n_in,
                              void* d_out, int out_size, void* d_ws, size_t ws_size,
                              hipStream_t stream) {
  (void)in_sizes; (void)n_in; (void)out_size; (void)ws_size;
  const float* ga      = (const float*)d_in[0];
  const float* vfin    = (const float*)d_in[1];
  const float* ex      = (const float*)d_in[2];
  const float* intr    = (const float*)d_in[3];
  const float* offs    = (const float*)d_in[4];
  const float* pg_w    = (const float*)d_in[5];
  const float* pg_b    = (const float*)d_in[6];
  const float* pg_lg   = (const float*)d_in[7];
  const float* pg_lb   = (const float*)d_in[8];
  const float* pv_w    = (const float*)d_in[9];
  const float* pv_b    = (const float*)d_in[10];
  const float* pv_lg   = (const float*)d_in[11];
  const float* pv_lb   = (const float*)d_in[12];
  const float* ipw     = (const float*)d_in[13];   // [2, 384, 128]
  const float* ipb     = (const float*)d_in[14];   // [2, 384]
  const float* ow      = (const float*)d_in[15];   // [2, 128, 128]
  const float* ob      = (const float*)d_in[16];   // [2, 128]
  const float* dca_g   = (const float*)d_in[17];
  const float* dca_b   = (const float*)d_in[18];
  const float* gate_w  = (const float*)d_in[19];   // [128, 256]
  const float* gate_b  = (const float*)d_in[20];
  const float* r_w1    = (const float*)d_in[21];   // [128, 256]
  const float* r_b1    = (const float*)d_in[22];
  const float* r_lg    = (const float*)d_in[23];
  const float* r_lb    = (const float*)d_in[24];
  const float* r_w2    = (const float*)d_in[25];   // [3, 128]
  const float* r_b2    = (const float*)d_in[26];
  const float* cent    = (const float*)d_in[27];   // [3, 64, 128]
  const float* cw      = (const float*)d_in[28];   // [3, 64, 128]
  const float* cb      = (const float*)d_in[29];   // [3, 64]
  const float* bn_w    = (const float*)d_in[30];   // [3, 256, 8192]
  const float* bn_b    = (const float*)d_in[31];
  const float* bn_lg   = (const float*)d_in[32];
  const float* bn_lb   = (const float*)d_in[33];
  float* out = (float*)d_out;

  char* wsb = (char*)d_ws;
  size_t off = 0;
  auto takeb = [&](size_t bytes) -> char* {
    char* p = wsb + off;
    off += (bytes + 255) & ~(size_t)255;
    return p;
  };
  float*     vfeat   = (float*)takeb((size_t)BN_ * C_ * 4);
  float*     g_emb   = (float*)takeb((size_t)BN_ * D_ * 4);
  float*     v_emb   = (float*)takeb((size_t)BN_ * D_ * 4);
  float*     tmp     = (float*)takeb((size_t)BN_ * D_ * 4);
  _Float16*  Qh      = (_Float16*)takeb((size_t)BN_ * D_ * 2);
  _Float16*  KVh     = (_Float16*)takeb((size_t)BN_ * 2 * D_ * 2);
  _Float16*  attnh   = (_Float16*)takeb((size_t)BN_ * D_ * 2);
  float*     fcat    = (float*)takeb((size_t)BN_ * 2 * D_ * 4);
  float*     gate    = (float*)takeb((size_t)BN_ * D_ * 4);
  float*     fused   = (float*)takeb((size_t)BN_ * D_ * 4);
  float*     xn      = (float*)takeb((size_t)BN_ * D_ * 4);
  float*     assg    = (float*)takeb((size_t)BN_ * K_ * 4);
  float*     asum    = (float*)takeb((size_t)B_ * K_ * 4);
  float*     pasum   = (float*)takeb((size_t)B_ * 32 * K_ * 4);
  float*     pmean   = (float*)takeb((size_t)B_ * 2 * 16 * D_ * 4);
  float*     vl      = (float*)takeb((size_t)B_ * K_ * D_ * 4);
  float*     descp   = (float*)takeb((size_t)16 * K_ * D_ * 4);   // padded to 16 rows
  float*     d2      = (float*)takeb((size_t)16 * OUT_ * 4);      // padded to 16 rows
  float*     descs   = (float*)takeb((size_t)3 * B_ * OUT_ * 4);
  float*     rip     = (float*)takeb((size_t)16 * 2 * D_ * 4);    // padded to 16 rows
  float*     rh      = (float*)takeb((size_t)B_ * D_ * 4);
  float*     rwv     = (float*)takeb(32 * 4);
  float*     ga_pack = (float*)takeb((size_t)BN_ * 32 * 4);
  float*     pgw_pk  = (float*)takeb((size_t)D_ * 32 * 4);

  dim3 wave(32);

  // pack K=14 operands to K=32, zero the M-padded buffers
  k_pack14<<<BN_, 32, 0, stream>>>(ga, ga_pack);
  k_pack14<<<D_, 32, 0, stream>>>(pg_w, pgw_pk);
  k_zero<<<(16 * 2 * D_ + 255) / 256, 256, 0, stream>>>(rip, 16 * 2 * D_);
  k_zero<<<(16 * K_ * D_ + 255) / 256, 256, 0, stream>>>(descp, 16 * K_ * D_);

  // projection + sampling
  k_proj<<<BN_, C_, 0, stream>>>(ga, vfin, ex, intr, offs, vfeat);

  // encoders
  k_gemm_t<float, float><<<dim3(D_ / 16, BN_ / 16), wave, 0, stream>>>(
      ga_pack, pgw_pk, pg_b, tmp, 32, 32, 32, D_, 0);
  k_ln_act<<<BN_, D_, 0, stream>>>(tmp, nullptr, pg_lg, pg_lb, g_emb, D_, D_, D_, 1);
  k_gemm_t<float, float><<<dim3(D_ / 16, BN_ / 16), wave, 0, stream>>>(
      vfeat, pv_w, pv_b, tmp, C_, C_, C_, D_, 0);
  k_ln_act<<<BN_, D_, 0, stream>>>(tmp, nullptr, pv_lg, pv_lb, v_emb, D_, D_, D_, 1);

  // two cross attentions
  for (int i = 0; i < 2; ++i) {
    const float* qs = i ? v_emb : g_emb;
    const float* ks = i ? g_emb : v_emb;
    const float* ipw_i = ipw + (size_t)i * 3 * D_ * D_;
    const float* ipb_i = ipb + (size_t)i * 3 * D_;
    k_gemm_t<float, _Float16><<<dim3(D_ / 16, BN_ / 16), wave, 0, stream>>>(
        qs, ipw_i, ipb_i, Qh, D_, D_, D_, D_, 0);
    k_gemm_t<float, _Float16><<<dim3((2 * D_) / 16, BN_ / 16), wave, 0, stream>>>(
        ks, ipw_i + (size_t)D_ * D_, ipb_i + D_, KVh, D_, D_, D_, 2 * D_, 0);
    k_flash<<<dim3(N_ / 16, B_ * 4), wave, 0, stream>>>(Qh, KVh, attnh);
    k_gemm_t<_Float16, float><<<dim3(D_ / 16, BN_ / 16), wave, 0, stream>>>(
        attnh, ow + (size_t)i * D_ * D_, ob + i * D_, tmp, D_, D_, D_, D_, 0);
    k_ln_act<<<BN_, D_, 0, stream>>>(tmp, qs, dca_g + i * D_, dca_b + i * D_,
                                     fcat + i * D_, D_, D_, 2 * D_, 0);
  }

  // gate + fuse
  k_gemm_t<float, float><<<dim3(D_ / 16, BN_ / 16), wave, 0, stream>>>(
      fcat, gate_w, gate_b, gate, 2 * D_, 2 * D_, 2 * D_, D_, 1);
  k_gate_fuse<<<BN_, D_, 0, stream>>>(fcat, gate, fused);

  // router (M padded to 16 rows, pad rows pre-zeroed)
  k_rowmean_p1<<<dim3(B_, 16, 2), D_, 0, stream>>>(g_emb, v_emb, pmean);
  k_rowmean_p2<<<dim3(B_, 2), D_, 0, stream>>>(pmean, rip);
  k_gemm_t<float, float><<<dim3(D_ / 16, 1), wave, 0, stream>>>(
      rip, r_w1, r_b1, tmp, 2 * D_, 2 * D_, 2 * D_, D_, 0);
  k_ln_act<<<B_, D_, 0, stream>>>(tmp, nullptr, r_lg, r_lb, rh, D_, D_, D_, 1);
  k_router<<<1, 32, 0, stream>>>(rh, r_w2, r_b2, rwv, out);

  // three NetVLAD branches
  const float* feats[3] = { g_emb, v_emb, fused };
  for (int i = 0; i < 3; ++i) {
    k_l2rows<<<BN_, D_, 0, stream>>>(feats[i], xn);
    k_gemm_t<float, float><<<dim3(K_ / 16, BN_ / 16), wave, 0, stream>>>(
        xn, cw + (size_t)i * K_ * D_, cb + i * K_, assg, D_, D_, D_, K_, 0);
    k_softmax64<<<BN_, K_, 0, stream>>>(assg);
    k_asum_p1<<<dim3(B_, 32), K_, 0, stream>>>(assg, pasum);
    k_asum_p2<<<B_, K_, 0, stream>>>(pasum, asum);
    k_vlad_gemm<<<dim3(D_ / 16, K_ / 16, B_), wave, 0, stream>>>(assg, xn, vl);
    k_vlad_post1<<<B_ * K_, D_, 0, stream>>>(vl, asum, cent + (size_t)i * K_ * D_, descp);
    k_vlad_post2<<<B_, 256, 0, stream>>>(descp);
    k_gemm_t<float, float><<<dim3(OUT_ / 16, 1), wave, 0, stream>>>(
        descp, bn_w + (size_t)i * OUT_ * K_ * D_, bn_b + i * OUT_, d2,
        K_ * D_, K_ * D_, K_ * D_, OUT_, 0);
    k_bn_post<<<B_, OUT_, 0, stream>>>(d2, bn_lg + i * OUT_, bn_lb + i * OUT_,
                                       descs + (size_t)i * B_ * OUT_);
  }

  k_final<<<B_, OUT_, 0, stream>>>(descs, rwv, out);
}
